// Mamba2Block_9088150798904
// MI455X (gfx1250) — matrix operand order repaired and checked by reading, not compiled, not run on hardware
//
#include <hip/hip_runtime.h>
#include <hip/hip_bf16.h>

typedef __bf16 bf16;
typedef __attribute__((ext_vector_type(16))) __bf16 bf16x16;
typedef __attribute__((ext_vector_type(8)))  __bf16 bf16x8;
typedef __attribute__((ext_vector_type(8)))  float  f32x8;

constexpr int   Bn = 4, S = 4096, H = 1024;
constexpr int   M  = Bn * S;            // 16384 rows
constexpr int   NC = 32;                // scan chunks
constexpr int   CL = S / NC;            // 128 steps per chunk

// ---------------- workspace layout (bytes) ----------------
constexpr size_t OFF_XB = 0;                                   // M*H bf16      = 32 MB
constexpr size_t OFF_WB = OFF_XB + (size_t)M * H * 2;          // 6*H*H bf16    = 12 MB
constexpr size_t OFF_A  = OFF_WB + (size_t)6 * H * H * 2;      // M*H f32       = 64 MB
constexpr size_t OFF_BX = OFF_A  + (size_t)M * H * 4;          // M*H f32       = 64 MB
constexpr size_t OFF_RG = OFF_BX + (size_t)M * H * 4;          // M*H bf16      = 32 MB
constexpr size_t OFF_SK = OFF_RG + (size_t)M * H * 2;          // M*H bf16      = 32 MB
constexpr size_t OFF_Y  = OFF_SK + (size_t)M * H * 2;          // M*H bf16      = 32 MB
constexpr size_t OFF_CA = OFF_Y  + (size_t)M * H * 2;          // B*NC*H f32
constexpr size_t OFF_CB = OFF_CA + (size_t)Bn * NC * H * 4;
constexpr size_t OFF_CS = OFF_CB + (size_t)Bn * NC * H * 4;

__device__ __forceinline__ float sigmoid_fast(float x) {
    return 1.0f / (1.0f + __expf(-x));
}
__device__ __forceinline__ float tanh_fast(float x) {
    return 2.0f * sigmoid_fast(2.0f * x) - 1.0f;
}

// ---------------- f32 -> bf16 conversion ----------------
__global__ void cvt_f32_bf16(const float* __restrict__ src, bf16* __restrict__ dst, int n) {
    int i = blockIdx.x * blockDim.x + threadIdx.x;
    int stride = gridDim.x * blockDim.x;
    for (; i < n; i += stride) dst[i] = (bf16)src[i];
}

// A-fragment loader: 16x32 bf16 tile, lane(half,m): K = {half*8..+7, 16+half*8..+7}
__device__ __forceinline__ bf16x16 load_afrag(const bf16* arow, int k0, int hf) {
    const bf16x8 lo = *reinterpret_cast<const bf16x8*>(arow + k0 + hf * 8);
    const bf16x8 hi = *reinterpret_cast<const bf16x8*>(arow + k0 + 16 + hf * 8);
    bf16x16 a;
    #pragma unroll
    for (int i = 0; i < 8; ++i) { a[i] = lo[i]; a[8 + i] = hi[i]; }
    return a;
}
// B-fragment loader: 32x16 bf16 tile, lane(half,n): K = {half*8..+7, 16+half*8..+7}
__device__ __forceinline__ bf16x16 load_bfrag(const bf16* brow, int k0, int hf) {
    const bf16x8 lo = *reinterpret_cast<const bf16x8*>(brow + k0 + hf * 8);
    const bf16x8 hi = *reinterpret_cast<const bf16x8*>(brow + k0 + 16 + hf * 8);
    bf16x16 b;
    #pragma unroll
    for (int i = 0; i < 8; ++i) { b[i] = lo[i]; b[8 + i] = hi[i]; }
    return b;
}

// ---------------- fused 5-way projection GEMM ----------------
// out[m,n] = sum_k x[m,k] * W[n,k]  for W in {W_in, W_a, W_b, W_c, W_d}
// Register blocking: each wave owns 4 M-tiles x 1 N-tile across all 5 weights:
// every B fragment feeds 4 WMMAs, every A fragment feeds 5 -> ~36 FLOP/L2-byte.
// Whole operand set (44 MB bf16) lives in the 192 MB L2, so no LDS staging.
__global__ void __launch_bounds__(128)
proj5_wmma(const bf16* __restrict__ xb, const bf16* __restrict__ wb,
           const float* __restrict__ b_a, const float* __restrict__ b_b,
           const float* __restrict__ b_c,
           float* __restrict__ aBuf, float* __restrict__ bxBuf,
           bf16* __restrict__ rgBuf, bf16* __restrict__ skBuf)
{
    const int lane  = threadIdx.x & 31;
    const int wave  = threadIdx.x >> 5;
    const int mBase = blockIdx.x * 64;                 // 4 m-tiles of 16
    const int nBase = blockIdx.y * 64 + wave * 16;     // 1 n-tile per wave
    const int hf    = lane >> 4;
    const int mr    = lane & 15;

    f32x8 acc[5][4];
    #pragma unroll
    for (int w = 0; w < 5; ++w)
        #pragma unroll
        for (int mt = 0; mt < 4; ++mt)
            acc[w][mt] = f32x8{0.f, 0.f, 0.f, 0.f, 0.f, 0.f, 0.f, 0.f};

    const bf16* arow[4];
    #pragma unroll
    for (int mt = 0; mt < 4; ++mt)
        arow[mt] = xb + (size_t)(mBase + mt * 16 + mr) * H;
    const bf16* brow[5];
    #pragma unroll
    for (int w = 0; w < 5; ++w)
        brow[w] = wb + (size_t)w * H * H + (size_t)(nBase + mr) * H;

    for (int k0 = 0; k0 < H; k0 += 32) {
        bf16x16 afrag[4];
        #pragma unroll
        for (int mt = 0; mt < 4; ++mt)
            afrag[mt] = load_afrag(arow[mt], k0, hf);
        #pragma unroll
        for (int w = 0; w < 5; ++w) {
            const bf16x16 bfrag = load_bfrag(brow[w], k0, hf);
            #pragma unroll
            for (int mt = 0; mt < 4; ++mt)
                acc[w][mt] = __builtin_amdgcn_wmma_f32_16x16x32_bf16(
                    false, afrag[mt], false, bfrag, (short)0, acc[w][mt], false, false);
        }
    }

    // Epilogue: fuse activations + write-term
    const int col = nBase + mr;
    const float ba = b_a[col], bb = b_b[col], bc = b_c[col];
    #pragma unroll
    for (int mt = 0; mt < 4; ++mt) {
        #pragma unroll
        for (int j = 0; j < 8; ++j) {
            const int row = mBase + mt * 16 + hf * 8 + j;   // C/D: m = vgpr + 8*half
            const size_t idx = (size_t)row * H + col;
            const float content = tanh_fast(acc[0][mt][j]);
            const float a       = sigmoid_fast(acc[1][mt][j] + ba);
            const float wg      = sigmoid_fast(acc[2][mt][j] + bb);
            const float rg      = sigmoid_fast(acc[3][mt][j] + bc);
            const float sk      = acc[4][mt][j];
            aBuf[idx]  = a;
            bxBuf[idx] = (1.0f - a) * wg * content;
            rgBuf[idx] = (bf16)rg;
            skBuf[idx] = (bf16)sk;
        }
    }
}

// ---------------- chunked linear-recurrence scan ----------------
__global__ void scan_chunk_reduce(const float* __restrict__ aBuf,
                                  const float* __restrict__ bxBuf,
                                  const unsigned char* __restrict__ mask,
                                  float* __restrict__ cA, float* __restrict__ cB)
{
    const int tid = blockIdx.x * blockDim.x + threadIdx.x;   // < B*NC*H
    const int h = tid % H;
    const int c = (tid / H) % NC;
    const int b = tid / (H * NC);
    float Ap = 1.f, Bg = 0.f;
    const int sBase = c * CL;
    for (int t = 0; t < CL; ++t) {
        const int s = sBase + t;
        const bool m = mask[b * S + s] != 0;
        const size_t idx = ((size_t)b * S + s) * H + h;
        const float a  = m ? aBuf[idx]  : 1.0f;
        const float bx = m ? bxBuf[idx] : 0.0f;
        Bg = a * Bg + bx;
        Ap *= a;
    }
    cA[tid] = Ap;   // layout [b][c][h] == tid
    cB[tid] = Bg;
}

__global__ void scan_chunks(const float* __restrict__ cA, const float* __restrict__ cB,
                            const float* __restrict__ state_in,
                            float* __restrict__ cState, float* __restrict__ final_state)
{
    const int tid = blockIdx.x * blockDim.x + threadIdx.x;   // < B*H
    const int h = tid % H;
    const int b = tid / H;
    float st = state_in[b * H + h];
    for (int c = 0; c < NC; ++c) {
        const size_t i = ((size_t)b * NC + c) * H + h;
        cState[i] = st;                 // exclusive prefix entering chunk c
        st = cA[i] * st + cB[i];
    }
    final_state[b * H + h] = st;        // inclusive over all chunks = final state
}

__global__ void scan_apply(const float* __restrict__ aBuf, const float* __restrict__ bxBuf,
                           const bf16* __restrict__ rgBuf, const bf16* __restrict__ skBuf,
                           const unsigned char* __restrict__ mask,
                           const float* __restrict__ cState, bf16* __restrict__ yBuf)
{
    const int tid = blockIdx.x * blockDim.x + threadIdx.x;   // < B*NC*H
    const int h = tid % H;
    const int c = (tid / H) % NC;
    const int b = tid / (H * NC);
    float st = cState[tid];
    const int sBase = c * CL;
    for (int t = 0; t < CL; ++t) {
        const int s = sBase + t;
        const bool m = mask[b * S + s] != 0;
        const size_t idx = ((size_t)b * S + s) * H + h;
        const float a  = m ? aBuf[idx]  : 1.0f;
        const float bx = m ? bxBuf[idx] : 0.0f;
        st = a * st + bx;
        const float y = m ? ((float)rgBuf[idx] * st + (float)skBuf[idx]) : 0.0f;
        yBuf[idx] = (bf16)y;
    }
}

// ---------------- output GEMM: out = y @ W_out^T (f32 result) ----------------
// Register blocking: 4 M-tiles x 2 N-tiles per wave (8 WMMAs per k-step).
__global__ void __launch_bounds__(128)
out_wmma(const bf16* __restrict__ yb, const bf16* __restrict__ wOut,
         float* __restrict__ out)
{
    const int lane  = threadIdx.x & 31;
    const int wave  = threadIdx.x >> 5;
    const int mBase = blockIdx.x * 64;                  // 4 m-tiles
    const int nBase = blockIdx.y * 128 + wave * 32;     // 2 n-tiles
    const int hf    = lane >> 4;
    const int mr    = lane & 15;

    f32x8 acc[2][4];
    #pragma unroll
    for (int nt = 0; nt < 2; ++nt)
        #pragma unroll
        for (int mt = 0; mt < 4; ++mt)
            acc[nt][mt] = f32x8{0.f, 0.f, 0.f, 0.f, 0.f, 0.f, 0.f, 0.f};

    const bf16* arow[4];
    #pragma unroll
    for (int mt = 0; mt < 4; ++mt)
        arow[mt] = yb + (size_t)(mBase + mt * 16 + mr) * H;
    const bf16* brow[2];
    #pragma unroll
    for (int nt = 0; nt < 2; ++nt)
        brow[nt] = wOut + (size_t)(nBase + nt * 16 + mr) * H;

    for (int k0 = 0; k0 < H; k0 += 32) {
        bf16x16 afrag[4];
        #pragma unroll
        for (int mt = 0; mt < 4; ++mt)
            afrag[mt] = load_afrag(arow[mt], k0, hf);
        #pragma unroll
        for (int nt = 0; nt < 2; ++nt) {
            const bf16x16 bfrag = load_bfrag(brow[nt], k0, hf);
            #pragma unroll
            for (int mt = 0; mt < 4; ++mt)
                acc[nt][mt] = __builtin_amdgcn_wmma_f32_16x16x32_bf16(
                    false, afrag[mt], false, bfrag, (short)0, acc[nt][mt], false, false);
        }
    }

    #pragma unroll
    for (int nt = 0; nt < 2; ++nt) {
        const int col = nBase + nt * 16 + mr;
        #pragma unroll
        for (int mt = 0; mt < 4; ++mt) {
            #pragma unroll
            for (int j = 0; j < 8; ++j) {
                const int row = mBase + mt * 16 + hf * 8 + j;
                out[(size_t)row * H + col] = acc[nt][mt][j];
            }
        }
    }
}

// ---------------- launch ----------------
extern "C" void kernel_launch(void* const* d_in, const int* in_sizes, int n_in,
                              void* d_out, int out_size, void* d_ws, size_t ws_size,
                              hipStream_t stream)
{
    const float* x      = (const float*)d_in[0];
    const float* state  = (const float*)d_in[1];
    const unsigned char* mask = (const unsigned char*)d_in[2];   // jnp.bool_ -> 1 byte
    const float* W_in   = (const float*)d_in[3];
    const float* W_a    = (const float*)d_in[4];
    const float* b_a    = (const float*)d_in[5];
    const float* W_b    = (const float*)d_in[6];
    const float* b_b    = (const float*)d_in[7];
    const float* W_c    = (const float*)d_in[8];
    const float* b_c    = (const float*)d_in[9];
    const float* W_d    = (const float*)d_in[10];
    const float* W_out  = (const float*)d_in[11];

    float* out         = (float*)d_out;            // [B,S,H] f32
    float* final_state = out + (size_t)M * H;      // [B,H]   f32 (tuple concat)

    char* ws = (char*)d_ws;
    bf16*  xb  = (bf16*)(ws + OFF_XB);
    bf16*  wb  = (bf16*)(ws + OFF_WB);   // 6 matrices: in,a,b,c,d,out
    float* aB  = (float*)(ws + OFF_A);
    float* bxB = (float*)(ws + OFF_BX);
    bf16*  rgB = (bf16*)(ws + OFF_RG);
    bf16*  skB = (bf16*)(ws + OFF_SK);
    bf16*  yB  = (bf16*)(ws + OFF_Y);
    float* cA  = (float*)(ws + OFF_CA);
    float* cB  = (float*)(ws + OFF_CB);
    float* cS  = (float*)(ws + OFF_CS);

    // f32 -> bf16 conversions
    cvt_f32_bf16<<<2048, 256, 0, stream>>>(x, xb, M * H);
    const float* Ws[6] = {W_in, W_a, W_b, W_c, W_d, W_out};
    for (int i = 0; i < 6; ++i)
        cvt_f32_bf16<<<1024, 256, 0, stream>>>(Ws[i], wb + (size_t)i * H * H, H * H);

    // fused 5-projection GEMM + activation epilogue
    dim3 gProj(M / 64, H / 64);     // (256, 16), 4 waves/block, 1 n-tile each
    proj5_wmma<<<gProj, 128, 0, stream>>>(xb, wb, b_a, b_b, b_c, aB, bxB, rgB, skB);

    // 3-phase chunked linear-recurrence scan
    scan_chunk_reduce<<<(Bn * NC * H) / 256, 256, 0, stream>>>(aB, bxB, mask, cA, cB);
    scan_chunks<<<(Bn * H) / 256, 256, 0, stream>>>(cA, cB, state, cS, final_state);
    scan_apply<<<(Bn * NC * H) / 256, 256, 0, stream>>>(aB, bxB, rgB, skB, mask, cS, yB);

    // output projection (4 M-tiles x 2 N-tiles per wave)
    dim3 gOut(M / 64, H / 128);     // (256, 8)
    out_wmma<<<gOut, 128, 0, stream>>>(yB, wb + (size_t)5 * H * H, out);
}
